// SelfAttentionLayer_5145370821412
// MI455X (gfx1250) — matrix operand order repaired and checked by reading, not compiled, not run on hardware
//
#include <hip/hip_runtime.h>
#include <hip/hip_bf16.h>

// ---------------------------------------------------------------------------
// Shapes (compile-time constants from the reference)
// ---------------------------------------------------------------------------
#define XSZ     1024
#define NHEADS  16
#define NBASIS  10
#define MAXLEN  4096
#define BATCH   8
#define TNEW    128
#define TPREV   4096
#define TTOT    (TPREV + TNEW)      // 4224
#define HDIM    (XSZ / NHEADS)      // 64
#define MROWS   (BATCH * TNEW)      // 1024 flat token rows
#define KVS     68                  // padded LDS row stride (floats) for KV tiles

typedef __attribute__((ext_vector_type(16))) __bf16 v16bf;
typedef __attribute__((ext_vector_type(8)))  float  v8f;

#define WMMA_BF16(a, b, c) \
    __builtin_amdgcn_wmma_f32_16x16x32_bf16(false, (a), false, (b), (short)0, (c), false, false)

// ---------------------------------------------------------------------------
// Tensor Data Mover (CDNA5): 2D tile DMA global->LDS, descriptor per ISA 08 §8
// This toolchain exposes the 6-arg builtin:
//   (uint32x4 g0, int32x8 g1, int32x4 g2, int32x4 g3, int32x8 extra, i32 cpol)
// ---------------------------------------------------------------------------
#if defined(__gfx1250__) && __has_builtin(__builtin_amdgcn_tensor_load_to_lds) && \
    __has_builtin(__builtin_amdgcn_s_wait_tensorcnt)
#define USE_TDM 1
#else
#define USE_TDM 0
#endif

#if USE_TDM
typedef unsigned int v4ui __attribute__((ext_vector_type(4)));
typedef int          v8si __attribute__((ext_vector_type(8)));
typedef int          v4si __attribute__((ext_vector_type(4)));

// pad after every 64 DWORDs (code 5) by 4 DWORDs (code 3): row stride 68 floats
#define TDM_PAD_68 ((1u << 20) | (5u << 22) | (3u << 25))

__device__ inline void tdm_load_2d(const void* gsrc, unsigned int lds_byte_off,
                                   unsigned int data_size_log2,
                                   unsigned int tile_d0, unsigned int tile_d1,
                                   unsigned int tensor_d0, unsigned int tensor_d1,
                                   unsigned long long stride0_elems,
                                   unsigned int pad_bits) {
    unsigned long long ga = (unsigned long long)(size_t)gsrc;
    v4ui g0;
    g0[0] = 1u;                                              // count=1, user D#
    g0[1] = lds_byte_off;                                    // bits 63:32
    g0[2] = (unsigned int)ga;                                // global_addr[31:0]
    g0[3] = (unsigned int)((ga >> 32) & 0x1ffffffull) | (2u << 30);  // addr[56:32] | type=2
    v8si g1;
    g1[0] = (int)((data_size_log2 << 16) | pad_bits);        // mask=0, data_size, pad cfg
    g1[1] = (int)(tensor_d0 << 16);                          // tensor_dim0[15:0] @bits63:48
    g1[2] = (int)((tensor_d0 >> 16) | (tensor_d1 << 16));    // td0[31:16] | td1[15:0]
    g1[3] = (int)((tensor_d1 >> 16) | (tile_d0 << 16));      // td1[31:16] | tile_dim0
    g1[4] = (int)(tile_d1 & 0xffffu);                        // tile_dim1, tile_dim2=0
    g1[5] = (int)(unsigned int)(stride0_elems & 0xffffffffull);   // dim0_stride[31:0]
    g1[6] = (int)(unsigned int)((stride0_elems >> 32) & 0xffffull);
    g1[7] = 0;
    v4si z4 = {0, 0, 0, 0};
    v8si z8 = {0, 0, 0, 0, 0, 0, 0, 0};
    __builtin_amdgcn_tensor_load_to_lds(g0, g1, z4, z4, z8, 0);
}
#endif

// ---- fragment packing helpers (16-bit WMMA operands, contiguous k order used here) -------------
// A (16x32, bf16): lane L, h=L>>4: elem e -> K = h*16 + e (contiguous per lane)
__device__ inline v16bf pack_a_bf16(const __bf16* row, int half) {
    v16bf a;
#pragma unroll
    for (int p = 0; p < 4; ++p) {
        int k0 = half * 16 + 2 * p;
        int k1 = half * 16 + 8 + 2 * p;
        a[2 * p]         = row[k0];
        a[2 * p + 1]     = row[k0 + 1];
        a[8 + 2 * p]     = row[k1];
        a[8 + 2 * p + 1] = row[k1 + 1];
    }
    return a;
}
// B (32x16): lane L, n=L&15, h=L>>4; elem e -> K = h*16 + e (contiguous per lane)
__device__ inline v16bf pack_b_bf16(const __bf16* colmajor) {
    v16bf b;
#pragma unroll
    for (int e = 0; e < 16; ++e) b[e] = colmajor[e];
    return b;
}
__device__ inline v16bf pack_b_f32(const float* colmajor) {
    v16bf b;
#pragma unroll
    for (int e = 0; e < 16; ++e) b[e] = (__bf16)colmajor[e];
    return b;
}
__device__ inline v16bf pack_b_f32_strided(const float* base, int stride) {
    v16bf b;
#pragma unroll
    for (int e = 0; e < 16; ++e) b[e] = (__bf16)base[e * stride];
    return b;
}

// ---------------------------------------------------------------------------
// 1) LayerNorm -> bf16
// ---------------------------------------------------------------------------
__global__ void ln_kernel(const float* __restrict__ X, const float* __restrict__ g,
                          const float* __restrict__ bb, __bf16* __restrict__ Xn) {
    __shared__ float s1[256];
    __shared__ float s2[256];
    const int row = blockIdx.x;
    const int tid = threadIdx.x;
    float a = 0.f, b = 0.f;
    for (int i = tid; i < XSZ; i += 256) {
        float x = X[(size_t)row * XSZ + i];
        a += x; b += x * x;
    }
    s1[tid] = a; s2[tid] = b;
    __syncthreads();
    for (int off = 128; off > 0; off >>= 1) {
        if (tid < off) { s1[tid] += s1[tid + off]; s2[tid] += s2[tid + off]; }
        __syncthreads();
    }
    const float mu   = s1[0] * (1.0f / XSZ);
    const float var  = s2[0] * (1.0f / XSZ) - mu * mu;
    const float rstd = rsqrtf(var + 1e-5f);
    for (int i = tid; i < XSZ; i += 256) {
        float x = X[(size_t)row * XSZ + i];
        Xn[(size_t)row * XSZ + i] = (__bf16)((x - mu) * rstd * g[i] + bb[i]);
    }
}

// ---------------------------------------------------------------------------
// 2) WMMA-bf16 GEMM: out = A(MxK bf16) * W(KxN fp32) [+bias] [+resid]
//    Block tile 64x64, 4 waves, K-step 32.
//    Double-buffered: A tile via TDM (overlapped, wait tensorcnt<=1), W manual.
// ---------------------------------------------------------------------------
__global__ void gemm_kernel(const __bf16* __restrict__ A, const float* __restrict__ W,
                            const float* __restrict__ bias, const float* __restrict__ resid,
                            float* __restrict__ outF, __bf16* __restrict__ outB,
                            int M, int N, int K) {
    __shared__ __bf16 As[2][64][32];   // [buf][m][k]
    __shared__ __bf16 Bt[2][64][32];   // [buf][n][k] (transposed, contiguous K)

    const int tid  = threadIdx.x;
    const int lane = tid & 31;
    const int w    = tid >> 5;
    const int m0   = blockIdx.y * 64;
    const int n0   = blockIdx.x * 64;
    const int mloc = lane & 15;
    const int half = lane >> 4;
    const int nl   = tid & 63;       // W staging: column within tile
    const int kg2  = tid >> 6;       // W staging: k-group base (0..1)

    v8f acc[4];
#pragma unroll
    for (int i = 0; i < 4; ++i) {
        v8f z = {0.f, 0.f, 0.f, 0.f, 0.f, 0.f, 0.f, 0.f};
        acc[i] = z;
    }

    auto stageA = [&](int p, int k0) {
#if USE_TDM
        if (w == 0)
            tdm_load_2d(A + (size_t)m0 * K + k0,
                        (unsigned int)(size_t)&As[p][0][0],
                        /*log2 bytes*/ 1, /*tile*/ 32, 64, /*tensor*/ 32, 64,
                        (unsigned long long)K, 0u);
#else
        int rl = tid >> 1, hh = tid & 1;
        *(uint4*)&As[p][rl][hh * 16] =
            *(const uint4*)(A + (size_t)(m0 + rl) * K + k0 + hh * 16);
#endif
    };
    auto stageW = [&](int p, int k0) {
#pragma unroll
        for (int it = 0; it < 4; ++it) {
            int kb = (kg2 + it * 2) * 4;
            float f0 = 0.f, f1 = 0.f, f2 = 0.f, f3 = 0.f;
            if (n0 + nl < N) {
                const float* wp = W + (size_t)(k0 + kb) * N + n0 + nl;
                f0 = wp[0];
                f1 = wp[(size_t)N];
                f2 = wp[(size_t)2 * N];
                f3 = wp[(size_t)3 * N];
            }
            Bt[p][nl][kb + 0] = (__bf16)f0;
            Bt[p][nl][kb + 1] = (__bf16)f1;
            Bt[p][nl][kb + 2] = (__bf16)f2;
            Bt[p][nl][kb + 3] = (__bf16)f3;
        }
    };

    const int NK = K / 32;
    stageA(0, 0);                       // prologue: fill buffer 0
    stageW(0, 0);

    for (int i = 0; i < NK; ++i) {
        const int p  = i & 1;
        const int k0 = i * 32;
        __syncthreads();                // buf[1-p] consumers (iter i-1) done
        if (i + 1 < NK) {               // stage next tile into the free buffer
            stageA(1 - p, k0 + 32);
            stageW(1 - p, k0 + 32);
        }
#if USE_TDM
        if (w == 0) {                   // drain current buffer's TDM, keep next in flight
            if (i + 1 < NK) __builtin_amdgcn_s_wait_tensorcnt(1);
            else            __builtin_amdgcn_s_wait_tensorcnt(0);
        }
#endif
        __syncthreads();

        v16bf af = pack_a_bf16(&As[p][w * 16 + mloc][0], half);
#pragma unroll
        for (int nt = 0; nt < 4; ++nt) {
            v16bf bf_ = pack_b_bf16(&Bt[p][nt * 16 + mloc][half * 16]);
            acc[nt] = WMMA_BF16(af, bf_, acc[nt]);
        }
    }

    // epilogue: C/D layout row = half*8 + r, col = lane&15
#pragma unroll
    for (int nt = 0; nt < 4; ++nt) {
        int col = n0 + nt * 16 + mloc;
        if (col >= N) continue;
        float bv = bias ? bias[col] : 0.f;
#pragma unroll
        for (int r = 0; r < 8; ++r) {
            int row = m0 + w * 16 + half * 8 + r;
            float v = acc[nt][r] + bv;
            if (resid) v += resid[(size_t)row * N + col];
            if (outF) outF[(size_t)row * N + col] = v;
            if (outB) outB[(size_t)row * N + col] = (__bf16)v;
        }
    }
}

// ---------------------------------------------------------------------------
// 3) Flash attention with relative-basis band bias.
//    grid = B*H blocks; 8 waves; wave w owns query rows [w*16, w*16+16).
//    KV tiles double-buffered via TDM (wait tensorcnt<=2 keeps next in flight),
//    fp32 in LDS with padded rows, converted to bf16 at fragment pack.
// ---------------------------------------------------------------------------
__global__ void attn_kernel(const float* __restrict__ Q,   const float* __restrict__ Kn,
                            const float* __restrict__ Vn,  const float* __restrict__ cK,
                            const float* __restrict__ cV,  const float* __restrict__ Rg,
                            const float* __restrict__ bnd, __bf16* __restrict__ Aout) {
    __shared__ float  Ksf[2][32][KVS];    // [buf][key][dim]
    __shared__ float  Vsf[2][32][KVS];    // [buf][key][dim]
    __shared__ float  Rsh[TNEW][NBASIS];
    __shared__ __bf16 Ps[8][16][32];      // per-wave P re-layout scratch

    const int tid  = threadIdx.x;
    const int lane = tid & 31;
    const int w    = tid >> 5;
    const int b    = blockIdx.x >> 4;
    const int h    = blockIdx.x & 15;
    const int m    = lane & 15;
    const int hf   = lane >> 4;

    for (int i = tid; i < TNEW * NBASIS; i += 256) {
        int q = i / NBASIS, n = i - q * NBASIS;
        Rsh[q][n] = Rg[(size_t)(b * TNEW + q) * (NBASIS * NHEADS) + h * NBASIS + n];
    }

    // Q fragments, pre-scaled by 1/sqrt(HDIM)
    const float* qptr = Q + ((size_t)(b * TNEW + w * 16 + m)) * XSZ + h * HDIM;
    v16bf qa0, qa1;
#pragma unroll
    for (int p = 0; p < 4; ++p) {
        int k0 = hf * 16 + 2 * p;
        int k1 = hf * 16 + 8 + 2 * p;
        qa0[2 * p]         = (__bf16)(qptr[k0] * 0.125f);
        qa0[2 * p + 1]     = (__bf16)(qptr[k0 + 1] * 0.125f);
        qa0[8 + 2 * p]     = (__bf16)(qptr[k1] * 0.125f);
        qa0[8 + 2 * p + 1] = (__bf16)(qptr[k1 + 1] * 0.125f);
        qa1[2 * p]         = (__bf16)(qptr[32 + k0] * 0.125f);
        qa1[2 * p + 1]     = (__bf16)(qptr[32 + k0 + 1] * 0.125f);
        qa1[8 + 2 * p]     = (__bf16)(qptr[32 + k1] * 0.125f);
        qa1[8 + 2 * p + 1] = (__bf16)(qptr[32 + k1 + 1] * 0.125f);
    }

    float m_i[8], l_i[8];
    v8f acc[4];
#pragma unroll
    for (int r = 0; r < 8; ++r) { m_i[r] = -1e30f; l_i[r] = 0.f; }
#pragma unroll
    for (int i = 0; i < 4; ++i) {
        v8f z = {0.f, 0.f, 0.f, 0.f, 0.f, 0.f, 0.f, 0.f};
        acc[i] = z;
    }

    // stage a 32-key KV block; blocks never straddle cache/new (TPREV % 32 == 0)
    auto kvstage = [&](int p, int j0) {
        const float* kb = (j0 < TPREV)
            ? cK + ((size_t)b * TPREV + j0) * XSZ + h * HDIM
            : Kn + ((size_t)b * TNEW + (j0 - TPREV)) * XSZ + h * HDIM;
        const float* vb = (j0 < TPREV)
            ? cV + ((size_t)b * TPREV + j0) * XSZ + h * HDIM
            : Vn + ((size_t)b * TNEW + (j0 - TPREV)) * XSZ + h * HDIM;
#if USE_TDM
        if (w == 0) {
            tdm_load_2d(kb, (unsigned int)(size_t)&Ksf[p][0][0], 2,
                        HDIM, 32, HDIM, 32, (unsigned long long)XSZ, TDM_PAD_68);
            tdm_load_2d(vb, (unsigned int)(size_t)&Vsf[p][0][0], 2,
                        HDIM, 32, HDIM, 32, (unsigned long long)XSZ, TDM_PAD_68);
        }
#else
        for (int s = tid; s < 32 * 16; s += 256) {   // 512 float4 slots each
            int key = s >> 4, d4 = (s & 15) * 4;
            *(float4*)&Ksf[p][key][d4] = *(const float4*)(kb + (size_t)key * XSZ + d4);
            *(float4*)&Vsf[p][key][d4] = *(const float4*)(vb + (size_t)key * XSZ + d4);
        }
#endif
    };

    const int NBLK = TTOT / 32;     // 132
    kvstage(0, 0);                  // prologue: fill buffer 0

    for (int i = 0; i < NBLK; ++i) {
        const int p  = i & 1;
        const int j0 = i * 32;
        __syncthreads();            // buf[1-p] consumers (iter i-1) done
        if (i + 1 < NBLK) kvstage(1 - p, j0 + 32);
#if USE_TDM
        if (w == 0) {               // drain current buffer's 2 TDMs, keep next 2 in flight
            if (i + 1 < NBLK) __builtin_amdgcn_s_wait_tensorcnt(2);
            else              __builtin_amdgcn_s_wait_tensorcnt(0);
        }
#endif
        __syncthreads();

        // S tiles: keys [j0, j0+16) and [j0+16, j0+32)
        v8f z = {0.f, 0.f, 0.f, 0.f, 0.f, 0.f, 0.f, 0.f};
        v8f s0 = z, s1 = z;
        s0 = WMMA_BF16(qa0, pack_b_f32(&Ksf[p][m][hf * 16]), s0);
        s0 = WMMA_BF16(qa1, pack_b_f32(&Ksf[p][m][32 + hf * 16]), s0);
        s1 = WMMA_BF16(qa0, pack_b_f32(&Ksf[p][16 + m][hf * 16]), s1);
        s1 = WMMA_BF16(qa1, pack_b_f32(&Ksf[p][16 + m][32 + hf * 16]), s1);

        // per-element: relative-basis extra + band mask
        float sv[2][8];
#pragma unroll
        for (int tile = 0; tile < 2; ++tile) {
#pragma unroll
            for (int r = 0; r < 8; ++r) {
                int qq = w * 16 + hf * 8 + r;
                int j  = j0 + tile * 16 + m;
                int d  = (TTOT - TNEW) + qq - j;     // 4096 + q - j
                float s = tile ? s1[r] : s0[r];
                if (d >= 0 && d < MAXLEN) {
                    float ex = 0.f;
#pragma unroll
                    for (int n = 0; n < NBASIS; ++n)
                        ex += Rsh[qq][n] * bnd[n * MAXLEN + d];
                    s += ex;
                } else {
                    s += -1e9f;
                }
                sv[tile][r] = s;
            }
        }

        // online softmax: row stats live at element r (row = hf*8+r)
        float alpha[8], lsum[8];
#pragma unroll
        for (int r = 0; r < 8; ++r) {
            float mx = fmaxf(sv[0][r], sv[1][r]);
#pragma unroll
            for (int off = 1; off < 16; off <<= 1)
                mx = fmaxf(mx, __shfl_xor(mx, off, 32));
            float mn = fmaxf(m_i[r], mx);
            alpha[r] = __expf(m_i[r] - mn);
            m_i[r] = mn;
            float p0 = __expf(sv[0][r] - mn);
            float p1 = __expf(sv[1][r] - mn);
            Ps[w][hf * 8 + r][m]      = (__bf16)p0;
            Ps[w][hf * 8 + r][16 + m] = (__bf16)p1;
            float ps = p0 + p1;
#pragma unroll
            for (int off = 1; off < 16; off <<= 1)
                ps += __shfl_xor(ps, off, 32);
            lsum[r] = ps;
        }
#pragma unroll
        for (int r = 0; r < 8; ++r) l_i[r] = l_i[r] * alpha[r] + lsum[r];
#pragma unroll
        for (int nt = 0; nt < 4; ++nt)
#pragma unroll
            for (int r = 0; r < 8; ++r) acc[nt][r] *= alpha[r];

        __syncthreads();   // Ps visible before re-read in A layout

        // P re-layout to A-fragment, then O += P * V
        v16bf pf = pack_a_bf16(&Ps[w][m][0], hf);
#pragma unroll
        for (int nt = 0; nt < 4; ++nt) {
            v16bf vbf = pack_b_f32_strided(&Vsf[p][hf * 16][nt * 16 + m], KVS);
            acc[nt] = WMMA_BF16(pf, vbf, acc[nt]);
        }
    }

    // normalize & emit merged-head A in bf16 (input to the proj GEMM)
#pragma unroll
    for (int nt = 0; nt < 4; ++nt) {
#pragma unroll
        for (int r = 0; r < 8; ++r) {
            int qq  = w * 16 + hf * 8 + r;
            int col = nt * 16 + m;
            float v = acc[nt][r] / l_i[r];
            Aout[((size_t)(b * TNEW + qq)) * XSZ + h * HDIM + col] = (__bf16)v;
        }
    }
}

// ---------------------------------------------------------------------------
// launch
// ---------------------------------------------------------------------------
extern "C" void kernel_launch(void* const* d_in, const int* in_sizes, int n_in,
                              void* d_out, int out_size, void* d_ws, size_t ws_size,
                              hipStream_t stream) {
    const float* X   = (const float*)d_in[0];
    const float* cK  = (const float*)d_in[1];
    const float* cV  = (const float*)d_in[2];
    const float* Wq  = (const float*)d_in[3];
    const float* bq  = (const float*)d_in[4];
    const float* Wk  = (const float*)d_in[5];
    const float* Wv  = (const float*)d_in[6];
    const float* Wp  = (const float*)d_in[7];
    const float* bp  = (const float*)d_in[8];
    const float* Wr  = (const float*)d_in[9];
    const float* br  = (const float*)d_in[10];
    const float* bnd = (const float*)d_in[11];
    const float* lg  = (const float*)d_in[12];
    const float* lb  = (const float*)d_in[13];
    float* out = (float*)d_out;

    char* ws = (char*)d_ws;
    size_t off = 0;
    __bf16* Xn = (__bf16*)(ws + off); off += (size_t)MROWS * XSZ * 2;
    float*  Qf = (float*)(ws + off);  off += (size_t)MROWS * XSZ * 4;
    float*  Kf = (float*)(ws + off);  off += (size_t)MROWS * XSZ * 4;
    float*  Vf = (float*)(ws + off);  off += (size_t)MROWS * XSZ * 4;
    float*  Rf = (float*)(ws + off);  off += (size_t)MROWS * 256 * 4;
    __bf16* Ab = (__bf16*)(ws + off); off += (size_t)MROWS * XSZ * 2;

    hipLaunchKernelGGL(ln_kernel, dim3(MROWS), dim3(256), 0, stream, X, lg, lb, Xn);

    dim3 bs(128);
    dim3 gsq(XSZ / 64, MROWS / 64);
    hipLaunchKernelGGL(gemm_kernel, gsq, bs, 0, stream,
                       Xn, Wq, bq, (const float*)nullptr, Qf, (__bf16*)nullptr,
                       MROWS, XSZ, XSZ);
    hipLaunchKernelGGL(gemm_kernel, gsq, bs, 0, stream,
                       Xn, Wk, (const float*)nullptr, (const float*)nullptr, Kf, (__bf16*)nullptr,
                       MROWS, XSZ, XSZ);
    hipLaunchKernelGGL(gemm_kernel, gsq, bs, 0, stream,
                       Xn, Wv, (const float*)nullptr, (const float*)nullptr, Vf, (__bf16*)nullptr,
                       MROWS, XSZ, XSZ);
    dim3 gsr((NBASIS * NHEADS + 63) / 64, MROWS / 64);
    hipLaunchKernelGGL(gemm_kernel, gsr, bs, 0, stream,
                       Xn, Wr, br, (const float*)nullptr, Rf, (__bf16*)nullptr,
                       MROWS, NBASIS * NHEADS, XSZ);

    hipLaunchKernelGGL(attn_kernel, dim3(BATCH * NHEADS), dim3(256), 0, stream,
                       Qf, Kf, Vf, cK, cV, Rf, bnd, Ab);

    hipLaunchKernelGGL(gemm_kernel, gsq, bs, 0, stream,
                       Ab, Wp, bp, X, out, (__bf16*)nullptr,
                       MROWS, XSZ, XSZ);

    (void)in_sizes; (void)n_in; (void)out_size; (void)ws_size; (void)off;
}
